// TransformerEncoderLayer_70076686402239
// MI455X (gfx1250) — matrix operand order repaired and checked by reading, not compiled, not run on hardware
//
#include <hip/hip_runtime.h>
#include <hip/hip_bf16.h>

typedef __attribute__((ext_vector_type(16))) _Float16 v16h;
typedef __attribute__((ext_vector_type(8)))  _Float16 v8h;
typedef __attribute__((ext_vector_type(8)))  float    v8f;

#define WMMA_F32_F16(a, b, c) \
  __builtin_amdgcn_wmma_f32_16x16x32_f16(false, (a), false, (b), (short)0, (c), false, false)

static __device__ __forceinline__ v16h pack16(v8h lo, v8h hi) {
  return __builtin_shufflevector(lo, hi, 0,1,2,3,4,5,6,7,8,9,10,11,12,13,14,15);
}
// A-fragment (16x32 f16): two contiguous 16B chunks at p and p+16.
static __device__ __forceinline__ v16h load_a(const _Float16* p) {
  return pack16(*(const v8h*)p, *(const v8h*)(p + 16));
}

// ---------------------------------------------------------------------------
// fp32 -> fp16 convert
// ---------------------------------------------------------------------------
__global__ void f32_to_f16_kernel(const float* __restrict__ in,
                                  _Float16* __restrict__ out, int n) {
  int i = blockIdx.x * blockDim.x + threadIdx.x;
  int stride = gridDim.x * blockDim.x;
  for (; i < n; i += stride) out[i] = (_Float16)in[i];
}

// ---------------------------------------------------------------------------
// [S,D] -> [D,S] fp16 transpose, LDS-tiled 64x64 so both global access
// directions are coalesced (padded LDS stride to dodge bank conflicts).
// ---------------------------------------------------------------------------
__global__ __launch_bounds__(256)
void transpose_kernel(const _Float16* __restrict__ in,
                      _Float16* __restrict__ out, int S, int D) {
  __shared__ _Float16 tile[64][72];
  const int tx = threadIdx.x & 63;
  const int ty = threadIdx.x >> 6;          // 0..3
  const int stiles = S >> 6;
  const int s0 = (blockIdx.x % stiles) << 6;
  const int c0 = (blockIdx.x / stiles) << 6;
#pragma unroll
  for (int r = ty; r < 64; r += 4)
    tile[r][tx] = in[(size_t)(s0 + r) * D + c0 + tx];
  __syncthreads();
#pragma unroll
  for (int r = ty; r < 64; r += 4)
    out[(size_t)(c0 + r) * S + s0 + tx] = tile[tx][r];
}

// ---------------------------------------------------------------------------
// C[M,N] = act(A[M,K] * W[N,K]^T + bias[N])   (torch Linear semantics)
// One wave computes a 32x64 tile: 2 A-fragments x 4 B-fragments = 8 WMMAs per
// K=32 step. Fragments are double-buffered (software pipeline): step k+32 is
// loaded before step k is consumed so v_wmma hides the (L2-resident) weight
// stream. A guarded prefetch runs two steps ahead.
// Per-lane layouts used here (A/B in contiguous k order; valid as both match):
//   A 16x32 f16: row=lane&15; elems=K[kb..kb+15], kb=(lane>>4)*16 (one 32B load)
//   B 32x16 f16: col=lane&15; elems=K[kb2..kb2+15], kb2=(lane>>4)*16 (one 32B load)
//   D 16x16 f32: col=lane&15; row=(lane>>4)*8+v
// ---------------------------------------------------------------------------
template <bool RELU>
__global__ __launch_bounds__(128)
void gemm_kernel(const _Float16* __restrict__ A, const _Float16* __restrict__ W,
                 const float* __restrict__ bias,
                 float* __restrict__ outF, _Float16* __restrict__ outH,
                 int M, int N, int K) {
  const int wave = (int)((blockIdx.x * blockDim.x + threadIdx.x) >> 5);
  const int lane = (int)(threadIdx.x & 31);
  const int lo = lane & 15, hi = lane >> 4;
  const int ntiles = N >> 6;
  const int mt = wave / ntiles;
  const int nt = wave % ntiles;
  const int colBase = nt << 6;

  const _Float16* A0 = A + (size_t)(mt * 32 + lo) * K + hi * 16;
  const _Float16* A1 = A0 + (size_t)16 * K;
  const _Float16* W0 = W + (size_t)(colBase +  0 + lo) * K + hi * 16;
  const _Float16* W1 = W + (size_t)(colBase + 16 + lo) * K + hi * 16;
  const _Float16* W2 = W + (size_t)(colBase + 32 + lo) * K + hi * 16;
  const _Float16* W3 = W + (size_t)(colBase + 48 + lo) * K + hi * 16;

  v8f c00 = {}, c01 = {}, c02 = {}, c03 = {};
  v8f c10 = {}, c11 = {}, c12 = {}, c13 = {};

  // prologue: load K-step 0
  v16h a0 = *(const v16h*)(A0), a1 = *(const v16h*)(A1);
  v16h b0 = *(const v16h*)(W0), b1 = *(const v16h*)(W1);
  v16h b2 = *(const v16h*)(W2), b3 = *(const v16h*)(W3);

  for (int k0 = 32; k0 < K; k0 += 32) {
    // issue next step's loads before consuming current fragments
    v16h na0 = *(const v16h*)(A0 + k0), na1 = *(const v16h*)(A1 + k0);
    v16h nb0 = *(const v16h*)(W0 + k0), nb1 = *(const v16h*)(W1 + k0);
    v16h nb2 = *(const v16h*)(W2 + k0), nb3 = *(const v16h*)(W3 + k0);
    if (k0 + 32 < K) {  // warm L2/WGP$ two steps ahead
      __builtin_prefetch(W0 + k0 + 32, 0, 0);
      __builtin_prefetch(W1 + k0 + 32, 0, 0);
      __builtin_prefetch(W2 + k0 + 32, 0, 0);
      __builtin_prefetch(W3 + k0 + 32, 0, 0);
    }
    c00 = WMMA_F32_F16(a0, b0, c00);
    c01 = WMMA_F32_F16(a0, b1, c01);
    c02 = WMMA_F32_F16(a0, b2, c02);
    c03 = WMMA_F32_F16(a0, b3, c03);
    c10 = WMMA_F32_F16(a1, b0, c10);
    c11 = WMMA_F32_F16(a1, b1, c11);
    c12 = WMMA_F32_F16(a1, b2, c12);
    c13 = WMMA_F32_F16(a1, b3, c13);
    a0 = na0; a1 = na1; b0 = nb0; b1 = nb1; b2 = nb2; b3 = nb3;
  }
  // epilogue K-step
  c00 = WMMA_F32_F16(a0, b0, c00);
  c01 = WMMA_F32_F16(a0, b1, c01);
  c02 = WMMA_F32_F16(a0, b2, c02);
  c03 = WMMA_F32_F16(a0, b3, c03);
  c10 = WMMA_F32_F16(a1, b0, c10);
  c11 = WMMA_F32_F16(a1, b1, c11);
  c12 = WMMA_F32_F16(a1, b2, c12);
  c13 = WMMA_F32_F16(a1, b3, c13);

  union { v8f v; float f[8]; } u[2][4];
  u[0][0].v = c00; u[0][1].v = c01; u[0][2].v = c02; u[0][3].v = c03;
  u[1][0].v = c10; u[1][1].v = c11; u[1][2].v = c12; u[1][3].v = c13;
  const int col0 = colBase + lo;
  const float bs[4] = { bias[col0], bias[col0 + 16], bias[col0 + 32], bias[col0 + 48] };
#pragma unroll
  for (int a = 0; a < 2; ++a) {
#pragma unroll
    for (int v = 0; v < 8; ++v) {
      const int r = mt * 32 + a * 16 + hi * 8 + v;
      const size_t base = (size_t)r * N + col0;
#pragma unroll
      for (int j = 0; j < 4; ++j) {
        float x = u[a][j].f[v] + bs[j];
        if (RELU) x = fmaxf(x, 0.f);
        if (outF) outF[base + j * 16] = x;
        if (outH) outH[base + j * 16] = (_Float16)x;
      }
    }
  }
}

// ---------------------------------------------------------------------------
// Flash attention (non-causal). One wave per (head, 16-query tile), streaming
// keys 32 at a time with online softmax. We compute S^T = K*Q^T so the
// softmax output P lands directly in the WMMA A-fragment layout (row = query
// = lane&15, elems = this lane's 16 key columns) with no cross-lane moves.
// V is pre-transposed ([D,S]) so its B-fragments are two contiguous 16B loads;
// K and V fragments are loaded together at the top of each chunk so the V
// stream overlaps the S^T WMMAs and the softmax VALU work.
// ---------------------------------------------------------------------------
__global__ __launch_bounds__(128)
void flash_attn_kernel(const _Float16* __restrict__ Q,
                       const _Float16* __restrict__ Kmat,
                       const _Float16* __restrict__ VT,   // [D, S]
                       _Float16* __restrict__ O,          // [S, D]
                       int S, int D) {
  const int wave = (int)((blockIdx.x * blockDim.x + threadIdx.x) >> 5);
  const int lane = (int)(threadIdx.x & 31);
  const int lo = lane & 15, hi = lane >> 4;
  const int qtiles = S >> 4;
  const int h = wave / qtiles;
  const int qt = wave % qtiles;
  const int hd = 64;

  // Q as B operand (col = query = lane&15, elems along Dh), two K=32 steps.
  const _Float16* qrow = Q + (size_t)(qt * 16 + lo) * D + h * hd + hi * 8;
  const v16h qb0 = load_a(qrow);
  const v16h qb1 = load_a(qrow + 32);

  float mrun = -1e30f, lrun = 0.f;
  v8f o0 = {}, o1 = {}, o2 = {}, o3 = {};
  const float scale = 0.125f;  // 1/sqrt(64)

  for (int kc = 0; kc < S; kc += 32) {
    // ---- issue all loads for this chunk up front (K A-frags + V B-frags) ----
    const _Float16* kr0 = Kmat + (size_t)(kc + lo) * D + h * hd;
    const _Float16* kr1 = kr0 + (size_t)16 * D;
    v16h ka00 = load_a(kr0 + hi * 8);
    v16h ka01 = load_a(kr0 + 32 + hi * 8);
    v16h ka10 = load_a(kr1 + hi * 8);
    v16h ka11 = load_a(kr1 + 32 + hi * 8);
    const _Float16* vt0 = VT + (size_t)(h * hd + lo) * S + kc + hi * 8;
    v16h vb0 = load_a(vt0);
    v16h vb1 = load_a(vt0 + (size_t)16 * S);
    v16h vb2 = load_a(vt0 + (size_t)32 * S);
    v16h vb3 = load_a(vt0 + (size_t)48 * S);

    const v8f z = {};
    v8f st0 = WMMA_F32_F16(ka00, qb0, z); st0 = WMMA_F32_F16(ka01, qb1, st0);
    v8f st1 = WMMA_F32_F16(ka10, qb0, z); st1 = WMMA_F32_F16(ka11, qb1, st1);

    // S^T D-layout: this lane holds query lo, keys kc + (frag?16:0) + hi*8 + v.
    union { v8f v; float f[8]; } s0, s1;
    s0.v = st0; s1.v = st1;
    float lm = -1e30f;
#pragma unroll
    for (int v = 0; v < 8; ++v) {
      s0.f[v] *= scale; s1.f[v] *= scale;
      lm = fmaxf(lm, fmaxf(s0.f[v], s1.f[v]));
    }
    lm = fmaxf(lm, __shfl_xor(lm, 16));        // combine key halves
    const float mnew = fmaxf(mrun, lm);
    const float corr = __expf(mrun - mnew);

    // P in A-fragment layout: elems[0..7] from frag0, elems[8..15] from frag1.
    float rs = 0.f;
    union { v16h v; _Float16 f[16]; } pa;
#pragma unroll
    for (int v = 0; v < 8; ++v) {
      const float p0 = __expf(s0.f[v] - mnew);
      const float p1 = __expf(s1.f[v] - mnew);
      rs += p0 + p1;
      pa.f[v] = (_Float16)p0;
      pa.f[v + 8] = (_Float16)p1;
    }
    rs += __shfl_xor(rs, 16);
    lrun = lrun * corr + rs;
    mrun = mnew;

    // Rescale running O (D-layout rows = query (lane>>4)*8+v): broadcast corr.
    union { v8f v; float f[8]; } a0, a1, a2, a3;
    a0.v = o0; a1.v = o1; a2.v = o2; a3.v = o3;
#pragma unroll
    for (int v = 0; v < 8; ++v) {
      const float cv = __shfl(corr, hi * 8 + v);
      a0.f[v] *= cv; a1.f[v] *= cv; a2.f[v] *= cv; a3.f[v] *= cv;
    }
    o0 = a0.v; o1 = a1.v; o2 = a2.v; o3 = a3.v;

    // O += P(16x32) * V(32x64)
    o0 = WMMA_F32_F16(pa.v, vb0, o0);
    o1 = WMMA_F32_F16(pa.v, vb1, o1);
    o2 = WMMA_F32_F16(pa.v, vb2, o2);
    o3 = WMMA_F32_F16(pa.v, vb3, o3);
  }

  const float linv = 1.f / lrun;
  union { v8f v; float f[8]; } a0, a1, a2, a3;
  a0.v = o0; a1.v = o1; a2.v = o2; a3.v = o3;
#pragma unroll
  for (int v = 0; v < 8; ++v) {
    const float lv = __shfl(linv, hi * 8 + v);
    const int r = qt * 16 + hi * 8 + v;
    const size_t base = (size_t)r * D + h * hd + lo;
    O[base]      = (_Float16)(a0.f[v] * lv);
    O[base + 16] = (_Float16)(a1.f[v] * lv);
    O[base + 32] = (_Float16)(a2.f[v] * lv);
    O[base + 48] = (_Float16)(a3.f[v] * lv);
  }
}

// ---------------------------------------------------------------------------
// out = LayerNorm(A + B) * g + b ; one 256-thread block per row, D = 1024.
// ---------------------------------------------------------------------------
__global__ __launch_bounds__(256)
void add_ln_kernel(const float* __restrict__ A, const float* __restrict__ B,
                   const float* __restrict__ g, const float* __restrict__ bb,
                   float* __restrict__ outF, _Float16* __restrict__ outH, int D) {
  const int row = blockIdx.x;
  const int tid = threadIdx.x;
  float vals[4];
  float s = 0.f, s2 = 0.f;
#pragma unroll
  for (int i = 0; i < 4; ++i) {
    const int c = tid + i * 256;
    const float v = A[(size_t)row * D + c] + B[(size_t)row * D + c];
    vals[i] = v; s += v; s2 += v * v;
  }
#pragma unroll
  for (int off = 16; off >= 1; off >>= 1) {
    s += __shfl_xor(s, off);
    s2 += __shfl_xor(s2, off);
  }
  __shared__ float ss[8], ss2[8];
  if ((tid & 31) == 0) { ss[tid >> 5] = s; ss2[tid >> 5] = s2; }
  __syncthreads();
  float St = 0.f, St2 = 0.f;
#pragma unroll
  for (int w = 0; w < 8; ++w) { St += ss[w]; St2 += ss2[w]; }
  const float mu = St / (float)D;
  const float var = St2 / (float)D - mu * mu;
  const float r = rsqrtf(var + 1e-5f);
#pragma unroll
  for (int i = 0; i < 4; ++i) {
    const int c = tid + i * 256;
    const float y = (vals[i] - mu) * r * g[c] + bb[c];
    outF[(size_t)row * D + c] = y;
    if (outH) outH[(size_t)row * D + c] = (_Float16)y;
  }
}

// ---------------------------------------------------------------------------
// Host orchestration
// ---------------------------------------------------------------------------
extern "C" void kernel_launch(void* const* d_in, const int* in_sizes, int n_in,
                              void* d_out, int out_size, void* d_ws, size_t ws_size,
                              hipStream_t stream) {
  (void)n_in; (void)out_size; (void)ws_size;
  const int D = 1024, DFF = 4096;
  const int S = in_sizes[0] / D;  // 2048

  const float* x    = (const float*)d_in[0];
  const float* qinw = (const float*)d_in[1];
  const float* qinb = (const float*)d_in[2];
  const float* kinw = (const float*)d_in[3];
  const float* kinb = (const float*)d_in[4];
  const float* vinw = (const float*)d_in[5];
  const float* vinb = (const float*)d_in[6];
  const float* outw = (const float*)d_in[7];
  const float* outb = (const float*)d_in[8];
  const float* f1w  = (const float*)d_in[9];
  const float* f1b  = (const float*)d_in[10];
  const float* f2w  = (const float*)d_in[11];
  const float* f2b  = (const float*)d_in[12];
  const float* n1g  = (const float*)d_in[13];
  const float* n1b  = (const float*)d_in[14];
  const float* n2g  = (const float*)d_in[15];
  const float* n2b  = (const float*)d_in[16];
  const float* qow  = (const float*)d_in[17];
  const float* qob  = (const float*)d_in[18];
  const float* kow  = (const float*)d_in[19];
  const float* kob  = (const float*)d_in[20];
  const float* vow  = (const float*)d_in[21];
  const float* vob  = (const float*)d_in[22];

  char* ws = (char*)d_ws;
  const size_t MB = 1ull << 20;
  _Float16* wscr = (_Float16*)(ws + 0);        // 8 MiB weight scratch (reused per GEMM)
  _Float16* xh   = (_Float16*)(ws + 8  * MB);  // 4 MiB  x fp16 (later reused for y fp16)
  _Float16* qh   = (_Float16*)(ws + 12 * MB);  // 4 MiB
  _Float16* kh   = (_Float16*)(ws + 16 * MB);  // 4 MiB
  _Float16* vT   = (_Float16*)(ws + 20 * MB);  // 4 MiB  V transposed [D,S]
  _Float16* vh   = (_Float16*)(ws + 24 * MB);  // 4 MiB  V fp16 (later reused for attn out)
  float*    t0   = (float*)   (ws + 28 * MB);  // 8 MiB  attn-proj (later reused for ffn2 out)
  float*    hF   = (float*)   (ws + 36 * MB);  // 8 MiB  h fp32
  _Float16* hH   = (_Float16*)(ws + 44 * MB);  // 4 MiB  h fp16
  _Float16* mid  = (_Float16*)(ws + 48 * MB);  // 16 MiB relu(ffn1) fp16
  _Float16* yH   = xh;                         // y fp16 reuses x fp16 region

  float* qn = (float*)d_out;
  float* kn = (float*)d_out + (size_t)S * D;
  float* vn = (float*)d_out + (size_t)2 * S * D;
  float* yF = (float*)d_out + (size_t)3 * S * D;

  auto cvt = [&](const float* src, _Float16* dst, int n) {
    f32_to_f16_kernel<<<1024, 256, 0, stream>>>(src, dst, n);
  };
  auto gemm = [&](const _Float16* A, const _Float16* W, const float* bias,
                  float* oF, _Float16* oH, int M, int N, int K, bool relu) {
    const int waves = (M / 32) * (N / 64);
    if (relu)
      gemm_kernel<true><<<dim3(waves / 4), dim3(128), 0, stream>>>(A, W, bias, oF, oH, M, N, K);
    else
      gemm_kernel<false><<<dim3(waves / 4), dim3(128), 0, stream>>>(A, W, bias, oF, oH, M, N, K);
  };

  // 1) in-projections (fp16 outputs feed attention)
  cvt(x, xh, S * D);
  cvt(qinw, wscr, D * D); gemm(xh, wscr, qinb, nullptr, qh, S, D, D, false);
  cvt(kinw, wscr, D * D); gemm(xh, wscr, kinb, nullptr, kh, S, D, D, false);
  cvt(vinw, wscr, D * D); gemm(xh, wscr, vinb, nullptr, vh, S, D, D, false);
  transpose_kernel<<<dim3((S / 64) * (D / 64)), dim3(256), 0, stream>>>(vh, vT, S, D);

  // 2) flash attention (attn output reuses V fp16 region)
  {
    const int waves = 16 * (S / 16);
    flash_attn_kernel<<<dim3(waves / 4), dim3(128), 0, stream>>>(qh, kh, vT, vh, S, D);
  }

  // 3) out-proj + residual + LN1
  cvt(outw, wscr, D * D); gemm(vh, wscr, outb, t0, nullptr, S, D, D, false);
  add_ln_kernel<<<dim3(S), dim3(256), 0, stream>>>(x, t0, n1g, n1b, hF, hH, D);

  // 4) FFN + residual + LN2
  cvt(f1w, wscr, DFF * D); gemm(hH, wscr, f1b, nullptr, mid, S, DFF, D, true);
  cvt(f2w, wscr, D * DFF); gemm(mid, wscr, f2b, t0, nullptr, S, D, DFF, false);
  add_ln_kernel<<<dim3(S), dim3(256), 0, stream>>>(hF, t0, n2g, n2b, yF, yH, D);

  // 5) out-projections for next layer
  cvt(qow, wscr, D * D); gemm(yH, wscr, qob, qn, nullptr, S, D, D, false);
  cvt(kow, wscr, D * D); gemm(yH, wscr, kob, kn, nullptr, S, D, D, false);
  cvt(vow, wscr, D * D); gemm(yH, wscr, vob, vn, nullptr, S, D, D, false);
}
